// QuestionDecoder_5712306504162
// MI455X (gfx1250) — matrix operand order repaired and checked by reading, not compiled, not run on hardware
//
#include <hip/hip_runtime.h>
#include <hip/hip_bf16.h>

// Model constants
#define B_   32
#define LC_  384
#define LQ_  64
#define EM_  768
#define H_   768
#define HC_  384
#define V_   30522
#define NEG_ (-10000.0f)

typedef __attribute__((ext_vector_type(16))) __bf16 bf16x16;
typedef __attribute__((ext_vector_type(8)))  float  f32x8;

__device__ __forceinline__ float sigf(float x) { return 1.0f / (1.0f + __expf(-x)); }
__device__ __forceinline__ float splus(float x) { return (x > 20.0f) ? x : log1pf(__expf(x)); }
__device__ __forceinline__ f32x8 zero8() { f32x8 z = {0.f,0.f,0.f,0.f,0.f,0.f,0.f,0.f}; return z; }

// ---------------------------------------------------------------------------
// Generic WMMA GEMM:  C[M,N] = A[M,K] * op(B) (+bias1(+bias2))
//   BT=true : B is W[N,K] row-major (computes A @ W^T, the `_dense` pattern)
//   BT=false: B is B[K,N] row-major (attention-apply pattern)
// fp32 operands converted to bf16 in-register, fp32 accumulate via
// v_wmma_f32_16x16x32_bf16.  Block = 8 waves -> 32(M) x 256(N) tile,
// each wave: 1 A-frag reused across 4 B-frags per K step.
// Fragment maps used here (A/B: contiguous k, not the ISA table order):
//   A 16x32 bf16: lane L: row=L&15, k = (L>>4)*16 + i
//   B 32x16 bf16: lane L: col=L&15, k = (L>>4)*16 + i
//   C 16x16 f32 : lane L: col=L&15, row = j + 8*(L>>4)
// ---------------------------------------------------------------------------
template <bool BT>
__global__ void __launch_bounds__(256)
gemm_wmma(const float* __restrict__ A, long long sA,
          const float* __restrict__ Bm, long long sB,
          float* __restrict__ C, long long sC,
          int M, int N, int K, int lda, int ldb, int ldc,
          const float* __restrict__ bias1, const float* __restrict__ bias2) {
  A  += (long long)blockIdx.z * sA;
  Bm += (long long)blockIdx.z * sB;
  C  += (long long)blockIdx.z * sC;
  const int lane = threadIdx.x & 31;
  const int wid  = threadIdx.x >> 5;
  const int sel  = lane >> 4;
  const int r    = lane & 15;
  const int m0 = blockIdx.y * 32 + (wid & 1) * 16;
  const int n0 = blockIdx.x * 256 + (wid >> 1) * 64;

  f32x8 acc[4];
#pragma unroll
  for (int t = 0; t < 4; ++t) acc[t] = zero8();

  int arow = m0 + r; if (arow >= M) arow = M - 1;
  const float* Arow = A + (long long)arow * lda;

  for (int kb = 0; kb < K; kb += 32) {
    if (kb + 32 < K) __builtin_prefetch(Arow + kb + 32 + sel * 8, 0, 1);
    bf16x16 af;
    {
      const float* p = Arow + kb + sel * 16;
#pragma unroll
      for (int i = 0; i < 16; ++i) af[i] = (__bf16)p[i];
    }
#pragma unroll
    for (int t = 0; t < 4; ++t) {
      int n = n0 + t * 16 + r; if (n >= N) n = N - 1;
      bf16x16 bfv;
      if (BT) {
        const float* p = Bm + (long long)n * ldb + kb + sel * 16;
#pragma unroll
        for (int i = 0; i < 16; ++i) bfv[i] = (__bf16)p[i];
      } else {
        const float* p = Bm + (long long)(kb + sel * 16) * ldb + n;
#pragma unroll
        for (int i = 0; i < 16; ++i) bfv[i] = (__bf16)p[(long long)i * ldb];
      }
      acc[t] = __builtin_amdgcn_wmma_f32_16x16x32_bf16(false, af, false, bfv,
                                                       (short)0, acc[t], false, false);
    }
  }
#pragma unroll
  for (int t = 0; t < 4; ++t) {
    int n = n0 + t * 16 + r;
    if (n < N) {
      float bb = 0.0f;
      if (bias1) bb += bias1[n];
      if (bias2) bb += bias2[n];
#pragma unroll
      for (int j = 0; j < 8; ++j) {
        int m = m0 + j + sel * 8;
        if (m < M) C[(long long)m * ldc + n] = acc[t][j] + bb;
      }
    }
  }
}

// ---------------------------------------------------------------------------
// Persistent LSTM recurrence over T steps.  Gate pre-projections
// (x@Wih^T + bih + bhh) were computed by gemm_wmma into `gates` as
// [B][T][4*HN].  This kernel runs h@Whh^T per step with WMMA:
//   - h kept in LDS as bf16 [32][HN]  (feeds A-fragments directly)
//   - c kept in registers in C-fragment layout
//   - gate columns for i/f/g/o are tile-aligned per wave -> gates combined
//     entirely in VGPRs.
// 24 waves (768 thr): each wave owns NU = HN/384 u-tiles x both m-tiles x 4 gates.
// blockIdx.x selects direction when bidir (0=fwd col 0, 1=bwd col HN, reversed t).
// ---------------------------------------------------------------------------
template <int HN>
__global__ void __launch_bounds__(768)
lstm_wmma(const float* __restrict__ gates0, const float* __restrict__ gates1,
          const float* __restrict__ Whh0, const float* __restrict__ Whh1,
          const float* __restrict__ ih, const float* __restrict__ ic,
          float* __restrict__ out, int ldo, int T, int bidir) {
  constexpr int NU = HN / 384;
  const int dir = blockIdx.x;
  const float* gates = dir ? gates1 : gates0;
  const float* Whh   = dir ? Whh1   : Whh0;
  const int colOff   = bidir ? dir * HN : 0;
  const int rev      = bidir && dir;

  __shared__ __bf16 hs[32][HN];
  for (int i = threadIdx.x; i < 32 * HN; i += blockDim.x) {
    int b = i / HN, u = i % HN;
    hs[b][u] = (__bf16)(ih ? ih[i] : 0.0f);
  }
  const int lane = threadIdx.x & 31;
  const int wv   = threadIdx.x >> 5;  // 0..23
  const int sel  = lane >> 4, r = lane & 15;

  float cst[NU][2][8];
#pragma unroll
  for (int i = 0; i < NU; ++i)
#pragma unroll
    for (int mt = 0; mt < 2; ++mt)
#pragma unroll
      for (int j = 0; j < 8; ++j) {
        int b = mt * 16 + j + 8 * sel;
        int u = (wv + i * 24) * 16 + r;
        cst[i][mt][j] = ic ? ic[b * HN + u] : 0.0f;
      }
  __syncthreads();

  for (int step = 0; step < T; ++step) {
    const int t = rev ? (T - 1 - step) : step;
    f32x8 acc[NU][2][4];
#pragma unroll
    for (int i = 0; i < NU; ++i)
#pragma unroll
      for (int mt = 0; mt < 2; ++mt)
#pragma unroll
        for (int q = 0; q < 4; ++q) acc[i][mt][q] = zero8();

    for (int kb = 0; kb < HN; kb += 32) {
      bf16x16 af[2];
#pragma unroll
      for (int mt = 0; mt < 2; ++mt) {
        int row = mt * 16 + r;
#pragma unroll
        for (int i = 0; i < 16; ++i) af[mt][i] = hs[row][kb + sel * 16 + i];
      }
#pragma unroll
      for (int i = 0; i < NU; ++i) {
        int u0 = (wv + i * 24) * 16;
#pragma unroll
        for (int q = 0; q < 4; ++q) {
          int n = q * HN + u0 + r;
          const float* p = Whh + (long long)n * HN + kb + sel * 16;
          bf16x16 bfv;
#pragma unroll
          for (int x = 0; x < 16; ++x) bfv[x] = (__bf16)p[x];
#pragma unroll
          for (int mt = 0; mt < 2; ++mt)
            acc[i][mt][q] = __builtin_amdgcn_wmma_f32_16x16x32_bf16(
                false, af[mt], false, bfv, (short)0, acc[i][mt][q], false, false);
        }
      }
    }
    __syncthreads();  // all hs reads done before overwriting h
#pragma unroll
    for (int i = 0; i < NU; ++i) {
      int u = (wv + i * 24) * 16 + r;
#pragma unroll
      for (int mt = 0; mt < 2; ++mt) {
#pragma unroll
        for (int j = 0; j < 8; ++j) {
          int b = mt * 16 + j + 8 * sel;
          const float* gp = gates + ((long long)b * T + t) * (4 * HN) + u;
          float gi = acc[i][mt][0][j] + gp[0];
          float gf = acc[i][mt][1][j] + gp[HN];
          float gg = acc[i][mt][2][j] + gp[2 * HN];
          float go = acc[i][mt][3][j] + gp[3 * HN];
          float c  = sigf(gf) * cst[i][mt][j] + sigf(gi) * tanhf(gg);
          cst[i][mt][j] = c;
          float h = sigf(go) * tanhf(c);
          hs[b][u] = (__bf16)h;
          out[((long long)b * T + t) * ldo + colOff + u] = h;
        }
      }
    }
    __syncthreads();  // h writes visible before next step's reads
  }
}

// ---------------------------------------------------------------------------
// Elementwise / small kernels
// ---------------------------------------------------------------------------
__global__ void embed_ctx_k(const float* __restrict__ E, const float* __restrict__ Atag,
                            const int* __restrict__ cIds, const int* __restrict__ aIds,
                            float* __restrict__ outp) {
  int row = blockIdx.x;  // B*LC
  long long id = (long long)cIds[row];
  long long at = (long long)aIds[row];
  const float* e = E + id * EM_;
  const float* a = Atag + at * EM_;
  float* o = outp + (long long)row * EM_;
  for (int i = threadIdx.x; i < EM_; i += blockDim.x) o[i] = e[i] + a[i];
}

__global__ void embed_q_k(const float* __restrict__ E, const int* __restrict__ qIds,
                          float* __restrict__ outp) {
  int row = blockIdx.x;  // B*LQ
  long long id = (long long)qIds[row];
  const float* e = E + id * EM_;
  float* o = outp + (long long)row * EM_;
  for (int i = threadIdx.x; i < EM_; i += blockDim.x) o[i] = e[i];
}

// masks logits in place (kept for copy mechanism), writes softmax weights
__global__ void mask_softmax_k(float* __restrict__ logits, float* __restrict__ wout,
                               const int* __restrict__ qIds, const int* __restrict__ kIds,
                               int Lq, int Lk) {
  int row = blockIdx.x;  // B*Lq
  int b = row / Lq, qi = row % Lq;
  float qm = (qIds[b * Lq + qi] != 0) ? 1.0f : 0.0f;
  float* rowp = logits + (long long)row * Lk;
  float* wp = wout + (long long)row * Lk;
  const int* kr = kIds + (long long)b * Lk;
  __shared__ float red[256];
  float mx = -3.0e38f;
  for (int k = threadIdx.x; k < Lk; k += blockDim.x) {
    float km = (kr[k] != 0) ? 1.0f : 0.0f;
    float v = rowp[k] + (1.0f - qm * km) * NEG_;
    rowp[k] = v;
    mx = fmaxf(mx, v);
  }
  red[threadIdx.x] = mx; __syncthreads();
  for (int s = blockDim.x / 2; s > 0; s >>= 1) {
    if (threadIdx.x < s) red[threadIdx.x] = fmaxf(red[threadIdx.x], red[threadIdx.x + s]);
    __syncthreads();
  }
  mx = red[0]; __syncthreads();
  float sum = 0.0f;
  for (int k = threadIdx.x; k < Lk; k += blockDim.x) sum += __expf(rowp[k] - mx);
  red[threadIdx.x] = sum; __syncthreads();
  for (int s = blockDim.x / 2; s > 0; s >>= 1) {
    if (threadIdx.x < s) red[threadIdx.x] += red[threadIdx.x + s];
    __syncthreads();
  }
  float inv = 1.0f / red[0];
  for (int k = threadIdx.x; k < Lk; k += blockDim.x) wp[k] = __expf(rowp[k] - mx) * inv;
}

// c_out = gate * tanh(fusedLin) + (1-gate) * c_out_old   (c_out_old in cc[:, :768])
__global__ void fuse_gate_k(const float* __restrict__ fl, const float* __restrict__ gl,
                            const float* __restrict__ cc, float* __restrict__ cof, long long n) {
  long long i = (long long)blockIdx.x * blockDim.x + threadIdx.x;
  if (i >= n) return;
  long long row = i / EM_;
  int e = (int)(i % EM_);
  float g = sigf(gl[i]);
  cof[i] = g * tanhf(fl[i]) + (1.0f - g) * cc[row * (2 * H_) + e];
}

__global__ void maxout_k(const float* __restrict__ h2, float* __restrict__ qmax, long long n) {
  long long i = (long long)blockIdx.x * blockDim.x + threadIdx.x;
  if (i >= n) return;
  long long row = i / EM_;
  int e = (int)(i % EM_);
  const float* p = h2 + row * (2 * EM_) + 2 * e;
  qmax[i] = fmaxf(p[0], p[1]);
}

// scatter-max of masked attention logits into vocab logits (copy mechanism)
__global__ void copy_scatter_k(float* __restrict__ outL, const float* __restrict__ attnL,
                               const int* __restrict__ cIds) {
  int bq = blockIdx.x;  // B*LQ
  int b = bq / LQ_;
  __shared__ int ids[LC_];
  __shared__ float vals[LC_];
  for (int k = threadIdx.x; k < LC_; k += blockDim.x) {
    ids[k] = cIds[b * LC_ + k];
    vals[k] = attnL[(long long)bq * LC_ + k];
  }
  __syncthreads();
  for (int k = threadIdx.x; k < LC_; k += blockDim.x) {
    int id = ids[k];
    float m = vals[k];
    bool first = true;
    for (int kk = 0; kk < LC_; ++kk) {
      if (ids[kk] == id) {
        m = fmaxf(m, vals[kk]);
        if (kk < k) first = false;
      }
    }
    m = fmaxf(m, NEG_);
    if (first && m > NEG_) outL[(long long)bq * V_ + id] += m;
  }
}

__global__ void means_k(const float* __restrict__ cof, const float* __restrict__ qmax,
                        const int* __restrict__ aIds, const int* __restrict__ qIds,
                        float* __restrict__ a_mean, float* __restrict__ q_mean) {
  int b = blockIdx.x;
  float asum = 0.0f, qlen = 0.0f;
  for (int t = 0; t < LC_; ++t) asum += (float)aIds[b * LC_ + t];
  for (int q = 0; q < LQ_; ++q) qlen += (qIds[b * LQ_ + q] != 0) ? 1.0f : 0.0f;
  for (int e = threadIdx.x; e < EM_; e += blockDim.x) {
    float sa = 0.0f;
    for (int t = 0; t < LC_; ++t)
      sa += cof[((long long)b * LC_ + t) * EM_ + e] * (float)aIds[b * LC_ + t];
    a_mean[b * EM_ + e] = sa / asum;
    float sq = 0.0f;
    for (int q = 0; q < LQ_; ++q) {
      float qm = (qIds[b * LQ_ + q] != 0) ? 1.0f : 0.0f;
      sq += qmax[((long long)b * LQ_ + q) * EM_ + e] * qm;
    }
    q_mean[b * EM_ + e] = sq / qlen;
  }
}

// res[0..31]=bil(q,a); res[32..63]=bil(q, roll(a)); res[64..95]=bil(roll(q), a)
__global__ void bilinear_k(const float* __restrict__ q_mean, const float* __restrict__ a_mean,
                           const float* __restrict__ W, const float* __restrict__ bias,
                           float* __restrict__ res) {
  int idx = blockIdx.x;
  int b = idx & 31;
  int kind = idx >> 5;
  int bq = (kind == 2) ? ((b + 31) & 31) : b;
  int ba = (kind == 1) ? ((b + 31) & 31) : b;
  const float* qv = q_mean + bq * EM_;
  const float* av = a_mean + ba * EM_;
  __shared__ float red[256];
  float part = 0.0f;
  for (int e = threadIdx.x; e < EM_; e += blockDim.x) {
    const float* wr = W + (long long)e * H_;
    float d = 0.0f;
    for (int h = 0; h < H_; ++h) d += wr[h] * av[h];
    part += qv[e] * d;
  }
  red[threadIdx.x] = part; __syncthreads();
  for (int s = blockDim.x / 2; s > 0; s >>= 1) {
    if (threadIdx.x < s) red[threadIdx.x] += red[threadIdx.x + s];
    __syncthreads();
  }
  if (threadIdx.x == 0) res[idx] = red[0] + bias[0];
}

__global__ void loss_k(const float* __restrict__ res, float* __restrict__ outLoss) {
  __shared__ float red[128];
  int t = threadIdx.x;
  float v = 0.0f;
  if (t < 32) { float x = res[t]; v = (splus(x) - x) / 32.0f; }
  else if (t < 96) { float x = res[t]; v = 0.5f * splus(x) / 64.0f; }
  red[t] = v; __syncthreads();
  for (int s = 64; s > 0; s >>= 1) {
    if (t < s) red[t] += red[t + s];
    __syncthreads();
  }
  if (t == 0) *outLoss = red[0];
}

// ---------------------------------------------------------------------------
// Host side
// ---------------------------------------------------------------------------
static void gemmL(bool bt, const float* A, long long sA, const float* Bm, long long sB,
                  float* C, long long sC, int M, int N, int K, int lda, int ldb, int ldc,
                  const float* b1, const float* b2, int batch, hipStream_t s) {
  dim3 g((N + 255) / 256, (M + 31) / 32, batch);
  if (bt)
    gemm_wmma<true><<<g, dim3(256), 0, s>>>(A, sA, Bm, sB, C, sC, M, N, K, lda, ldb, ldc, b1, b2);
  else
    gemm_wmma<false><<<g, dim3(256), 0, s>>>(A, sA, Bm, sB, C, sC, M, N, K, lda, ldb, ldc, b1, b2);
}

extern "C" void kernel_launch(void* const* d_in, const int* in_sizes, int n_in,
                              void* d_out, int out_size, void* d_ws, size_t ws_size,
                              hipStream_t stream) {
  (void)in_sizes; (void)n_in; (void)out_size; (void)ws_size;
  auto F = [&](int i) { return (const float*)d_in[i]; };
  auto Ii = [&](int i) { return (const int*)d_in[i]; };

  // params dict insertion order (per setup_inputs):
  // l in {0,1}: ctx{l}f_{Wih,Whh,bih,bhh}, ctx{l}b_{...}, q{l}_{Wih,Whh,bih,bhh}
  const float *ctxWih[2][2], *ctxWhh[2][2], *ctxBih[2][2], *ctxBhh[2][2];
  const float *qWih[2], *qWhh[2], *qBih[2], *qBhh[2];
  for (int l = 0; l < 2; ++l) {
    for (int d = 0; d < 2; ++d) {
      int base = l * 12 + d * 4;
      ctxWih[l][d] = F(base + 0);
      ctxWhh[l][d] = F(base + 1);
      ctxBih[l][d] = F(base + 2);
      ctxBhh[l][d] = F(base + 3);
    }
    int base = l * 12 + 8;
    qWih[l] = F(base + 0); qWhh[l] = F(base + 1);
    qBih[l] = F(base + 2); qBhh[l] = F(base + 3);
  }
  const float* ctxlin_W = F(24); const float* ctxlin_b = F(25);
  const float* fusion_W = F(26); const float* gate_W = F(27);
  const float* qlin_W = F(28);   const float* qlin_b = F(29);
  const float* cat1_W = F(30);   const float* cat1_b = F(31);
  const float* cat2_W = F(32);   const float* cat2_b = F(33);
  const float* disc_W = F(34);   const float* disc_b = F(35);
  const float* E = F(36);        const float* Atag = F(37);
  const float* init_h = F(38);   const float* init_c = F(39);
  const int* c_ids = Ii(40);     const int* q_ids = Ii(41);
  const int* a_ids = Ii(42);

  // workspace bump allocator (fp32)
  char* wsb = (char*)d_ws;
  size_t off = 0;
  auto alloc = [&](size_t nf) { float* p = (float*)(wsb + off); off += nf * sizeof(float); return p; };
  const long long BLc = (long long)B_ * LC_, BLq = (long long)B_ * LQ_;
  float* cx   = alloc(BLc * EM_);       // layer-0 ctx input
  float* cx2  = alloc(BLc * EM_);       // layer-0 ctx output / layer-1 input
  float* gF   = alloc(BLc * 4 * HC_);   // fwd gates pre; later reused as fusedLin
  float* gB   = alloc(BLc * 4 * HC_);   // bwd gates pre; later reused as gateLin
  float* cc   = alloc(BLc * 2 * H_);    // [c_out | c_att] concat
  float* projC= alloc(BLc * H_);
  float* attC = alloc(BLc * LC_);       // self-attn logits -> weights in place
  float* cof  = alloc(BLc * H_);        // fused/gated c_out
  float* qx   = alloc(BLq * EM_);
  float* qg   = alloc(BLq * 4 * H_);    // q gates pre (reused across layers)
  float* qo0  = alloc(BLq * H_);
  float* qcat = alloc(BLq * 2 * H_);    // [q_out | c_att_q]
  float* projQ= alloc(BLq * H_);
  float* attQ = alloc(BLq * LC_);       // masked logits (kept for copy)
  float* attQW= alloc(BLq * LC_);       // softmax weights
  float* h1   = alloc(BLq * 2 * H_);
  float* h2   = alloc(BLq * 2 * H_);
  float* qmaxB= alloc(BLq * EM_);
  float* am   = alloc((size_t)B_ * EM_);
  float* qm   = alloc((size_t)B_ * EM_);
  float* bres = alloc(96);

  float* logits = (float*)d_out;                       // [B,LQ,V]
  float* lossp  = logits + (size_t)B_ * LQ_ * V_;      // scalar

  // ---- context encoder ----
  embed_ctx_k<<<B_ * LC_, 256, 0, stream>>>(E, Atag, c_ids, a_ids, cx);
  // layer 0
  gemmL(true, cx, 0, ctxWih[0][0], 0, gF, 0, B_*LC_, 4*HC_, EM_, EM_, EM_, 4*HC_,
        ctxBih[0][0], ctxBhh[0][0], 1, stream);
  gemmL(true, cx, 0, ctxWih[0][1], 0, gB, 0, B_*LC_, 4*HC_, EM_, EM_, EM_, 4*HC_,
        ctxBih[0][1], ctxBhh[0][1], 1, stream);
  lstm_wmma<384><<<2, 768, 0, stream>>>(gF, gB, ctxWhh[0][0], ctxWhh[0][1],
                                        nullptr, nullptr, cx2, H_, LC_, 1);
  // layer 1 (din = 2*HC = 768)
  gemmL(true, cx2, 0, ctxWih[1][0], 0, gF, 0, B_*LC_, 4*HC_, 2*HC_, 2*HC_, 2*HC_, 4*HC_,
        ctxBih[1][0], ctxBhh[1][0], 1, stream);
  gemmL(true, cx2, 0, ctxWih[1][1], 0, gB, 0, B_*LC_, 4*HC_, 2*HC_, 2*HC_, 2*HC_, 4*HC_,
        ctxBih[1][1], ctxBhh[1][1], 1, stream);
  lstm_wmma<384><<<2, 768, 0, stream>>>(gF, gB, ctxWhh[1][0], ctxWhh[1][1],
                                        nullptr, nullptr, cc, 2*H_, LC_, 1);
  // ctx self-attention: proj = c_out @ ctxlin^T + b
  gemmL(true, cc, 0, ctxlin_W, 0, projC, 0, B_*LC_, H_, H_, 2*H_, H_, H_,
        ctxlin_b, nullptr, 1, stream);
  // logits[b] = proj[b] @ c_out[b]^T   (batched)
  gemmL(true, projC, (long long)LC_*H_, cc, (long long)LC_*2*H_, attC, (long long)LC_*LC_,
        LC_, LC_, H_, H_, 2*H_, LC_, nullptr, nullptr, B_, stream);
  mask_softmax_k<<<B_ * LC_, 256, 0, stream>>>(attC, attC, c_ids, c_ids, LC_, LC_);
  // c_att[b] = w[b] @ c_out[b]  -> into cc[:, 768:]
  gemmL(false, attC, (long long)LC_*LC_, cc, (long long)LC_*2*H_, cc + H_, (long long)LC_*2*H_,
        LC_, H_, LC_, LC_, 2*H_, 2*H_, nullptr, nullptr, B_, stream);
  // fusion gate (reuse gF/gB as fusedLin/gateLin)
  gemmL(true, cc, 0, fusion_W, 0, gF, 0, B_*LC_, H_, 2*H_, 2*H_, 2*H_, H_,
        nullptr, nullptr, 1, stream);
  gemmL(true, cc, 0, gate_W, 0, gB, 0, B_*LC_, H_, 2*H_, 2*H_, 2*H_, H_,
        nullptr, nullptr, 1, stream);
  {
    long long n = BLc * EM_;
    fuse_gate_k<<<(int)((n + 255) / 256), 256, 0, stream>>>(gF, gB, cc, cof, n);
  }

  // ---- question LSTM ----
  embed_q_k<<<B_ * LQ_, 256, 0, stream>>>(E, q_ids, qx);
  gemmL(true, qx, 0, qWih[0], 0, qg, 0, B_*LQ_, 4*H_, EM_, EM_, EM_, 4*H_,
        qBih[0], qBhh[0], 1, stream);
  lstm_wmma<768><<<1, 768, 0, stream>>>(qg, qg, qWhh[0], qWhh[0],
                                        init_h, init_c, qo0, H_, LQ_, 0);
  gemmL(true, qo0, 0, qWih[1], 0, qg, 0, B_*LQ_, 4*H_, H_, H_, H_, 4*H_,
        qBih[1], qBhh[1], 1, stream);
  lstm_wmma<768><<<1, 768, 0, stream>>>(qg, qg, qWhh[1], qWhh[1],
                                        init_h + (size_t)B_*H_, init_c + (size_t)B_*H_,
                                        qcat, 2*H_, LQ_, 0);
  // q->c attention
  gemmL(true, qcat, 0, qlin_W, 0, projQ, 0, B_*LQ_, H_, H_, 2*H_, H_, H_,
        qlin_b, nullptr, 1, stream);
  gemmL(true, projQ, (long long)LQ_*H_, cof, (long long)LC_*H_, attQ, (long long)LQ_*LC_,
        LQ_, LC_, H_, H_, H_, LC_, nullptr, nullptr, B_, stream);
  mask_softmax_k<<<B_ * LQ_, 256, 0, stream>>>(attQ, attQW, q_ids, c_ids, LQ_, LC_);
  gemmL(false, attQW, (long long)LQ_*LC_, cof, (long long)LC_*H_, qcat + H_, (long long)LQ_*2*H_,
        LQ_, H_, LC_, LC_, H_, 2*H_, nullptr, nullptr, B_, stream);
  // cat1 -> cat2 -> maxout
  gemmL(true, qcat, 0, cat1_W, 0, h1, 0, B_*LQ_, 2*H_, 2*H_, 2*H_, 2*H_, 2*H_,
        cat1_b, nullptr, 1, stream);
  gemmL(true, h1, 0, cat2_W, 0, h2, 0, B_*LQ_, 2*EM_, 2*H_, 2*H_, 2*H_, 2*EM_,
        cat2_b, nullptr, 1, stream);
  {
    long long n = BLq * EM_;
    maxout_k<<<(int)((n + 255) / 256), 256, 0, stream>>>(h2, qmaxB, n);
  }
  // tied-embedding output GEMM (dominant: 2048x30522x768) straight into d_out
  gemmL(true, qmaxB, 0, E, 0, logits, 0, B_*LQ_, V_, EM_, EM_, EM_, V_,
        nullptr, nullptr, 1, stream);
  // copy mechanism (scatter-max of masked attn logits)
  copy_scatter_k<<<B_ * LQ_, 384, 0, stream>>>(logits, attQ, c_ids);
  // info loss
  means_k<<<B_, 256, 0, stream>>>(cof, qmaxB, a_ids, q_ids, am, qm);
  bilinear_k<<<96, 256, 0, stream>>>(qm, am, disc_W, disc_b, bres);
  loss_k<<<1, 128, 0, stream>>>(bres, lossp);
}
